// ConvNetLayer_16836271800759
// MI455X (gfx1250) — matrix operand order repaired and checked by reading, not compiled, not run on hardware
//
#include <hip/hip_runtime.h>
#include <hip/hip_bf16.h>

typedef __attribute__((ext_vector_type(16))) _Float16 v16h;
typedef __attribute__((ext_vector_type(8)))  _Float16 v8h;
typedef __attribute__((ext_vector_type(8)))  float    v8f;
typedef __attribute__((ext_vector_type(4)))  float    v4f;

#define Nn 4096
#define Dd 128
#define Bb 4

// ---------------------------------------------------------------------------
// Tiled f32 -> f16 transpose: out[c][r] = (f16) in[r][c], per batch slice.
// Used once for xT (4 MB) and UT (32 KB); cost is negligible vs the 268 MB
// adjacency stream.
// ---------------------------------------------------------------------------
__global__ __launch_bounds__(256)
void transpose_f32_to_f16(const float* __restrict__ in, _Float16* __restrict__ out,
                          int rows, int cols) {
    __shared__ float tile[32][33];
    const int b  = blockIdx.z;
    const int c0 = blockIdx.x * 32;
    const int r0 = blockIdx.y * 32;
    const float* src = in  + (size_t)b * rows * cols;
    _Float16*    dst = out + (size_t)b * rows * cols;
    for (int rr = threadIdx.y; rr < 32; rr += blockDim.y)
        tile[rr][threadIdx.x] = src[(size_t)(r0 + rr) * cols + c0 + threadIdx.x];
    __syncthreads();
    for (int rr = threadIdx.y; rr < 32; rr += blockDim.y)
        dst[(size_t)(c0 + rr) * rows + r0 + threadIdx.x] =
            (_Float16)tile[threadIdx.x][rr];
}

// ---------------------------------------------------------------------------
// Fused kernel: one wave per 16-wide i-tile.
//   Stage 1: AGG^T = X^T * mask(adj), K-loop over j (N=4096, step 32),
//            deg accumulated from the same adj loads.
//   Stage 2: OUT^T = U^T * AGG^T, scaled by 1/deg, relu, f32 store.
// ---------------------------------------------------------------------------
__global__ __launch_bounds__(256)
void gcn_fused(const float* __restrict__ adj, const _Float16* __restrict__ xT,
               const _Float16* __restrict__ UT, float* __restrict__ out) {
    __shared__ _Float16 lds_agg[8][16 * Dd];   // 32 KB: wave-private [i_local][d]

    const int lane = threadIdx.x & 31;
    const int wid  = threadIdx.x >> 5;
    const int tile = blockIdx.x * 8 + wid;     // 0..1023
    const int b    = tile >> 8;                // 256 i-tiles per batch
    const int i0   = (tile & 255) << 4;
    const int n    = lane & 15;
    const bool hi  = lane >= 16;

    const float*    adjB = adj + (size_t)b * Nn * Nn;   // [j][i]
    const _Float16* xTb  = xT  + (size_t)b * Dd * Nn;   // [d][j]

    v8f acc[8] = {};      // AGG^T fragments: d-tile t, rows d, cols i
    float dsum = 0.0f;    // per-lane partial of deg[i0+n]

    for (int j0 = 0; j0 < Nn; j0 += 32) {
        // ---- B fragment: mask tile, lane n = column i0+n, K = j ----
        const float* ap = adjB + (size_t)(j0 + (hi ? 16 : 0)) * Nn + i0 + n;
        if (j0 + 32 < Nn)
            __builtin_prefetch(ap + (size_t)32 * Nn, 0, 1);   // global_prefetch_b8
        union { v16h v; _Float16 e[16]; } bfrag;
        #pragma unroll
        for (int k = 0; k < 16; ++k) {
            float a = ap[(size_t)k * Nn];
            dsum += a;                                   // deg uses raw adj values
            bfrag.e[k] = (a > 0.0f) ? (_Float16)1.0f : (_Float16)0.0f;
        }
        // ---- A fragments: X^T rows d (lane n), K = j contiguous in xT ----
        #pragma unroll
        for (int t = 0; t < 8; ++t) {
            union { v16h v; v8h h[2]; } afrag;
            const _Float16* xp = xTb + (size_t)((t << 4) + n) * Nn + j0 + (hi ? 16 : 0);
            afrag.h[0] = *(const v8h*)(xp);        // K {0..7} / {16..23}
            afrag.h[1] = *(const v8h*)(xp + 8);    // K {8..15} / {24..31}
            acc[t] = __builtin_amdgcn_wmma_f32_16x16x32_f16(
                false, afrag.v, false, bfrag.v, (short)0, acc[t], false, false);
        }
    }

    // deg[i0+n] = lane half (j even-16s) + other half (j odd-16s)
    dsum += __shfl_xor(dsum, 16, 32);
    const float rdeg = 1.0f / dsum;

    // ---- hand AGG^T to stage 2 through wave-private LDS as [i][d] f16 ----
    _Float16* lbuf = &lds_agg[wid][0];
    #pragma unroll
    for (int t = 0; t < 8; ++t) {
        #pragma unroll
        for (int r = 0; r < 8; ++r) {
            int d = (t << 4) + r + (hi ? 8 : 0);       // C layout: VGPR r -> row
            lbuf[n * Dd + d] = (_Float16)acc[t][r];
        }
    }

    // ---- Stage 2: OUT^T[e][i] = sum_d U^T[e][d] * AGG^T[d][i] ----
    #pragma unroll 1
    for (int e0 = 0; e0 < Dd; e0 += 16) {
        v8f c2 = {};
        #pragma unroll
        for (int kk = 0; kk < Dd; kk += 32) {
            union { v16h v; v8h h[2]; } a2;    // U^T tile: rows e, K = d
            const _Float16* up = UT + (size_t)(e0 + n) * Dd + kk + (hi ? 16 : 0);
            a2.h[0] = *(const v8h*)(up);
            a2.h[1] = *(const v8h*)(up + 8);
            union { v16h v; v8h h[2]; } b2;    // AGG^T tile: lane n = col i, K = d
            const _Float16* lp = lbuf + n * Dd + kk + (hi ? 16 : 0);
            b2.h[0] = *(const v8h*)(lp);
            b2.h[1] = *(const v8h*)(lp + 8);
            c2 = __builtin_amdgcn_wmma_f32_16x16x32_f16(
                false, a2.v, false, b2.v, (short)0, c2, false, false);
        }
        // epilogue: lane holds out[b][i0+n][e0 + (hi?8:0) + 0..7] contiguous
        float vout[8];
        #pragma unroll
        for (int r = 0; r < 8; ++r)
            vout[r] = fmaxf(c2[r] * rdeg, 0.0f);
        float* op = out + (size_t)(b * Nn + i0 + n) * Dd + e0 + (hi ? 8 : 0);
        *(v4f*)op       = *(v4f*)&vout[0];
        *(v4f*)(op + 4) = *(v4f*)&vout[4];
    }
}

extern "C" void kernel_launch(void* const* d_in, const int* in_sizes, int n_in,
                              void* d_out, int out_size, void* d_ws, size_t ws_size,
                              hipStream_t stream) {
    const float* x   = (const float*)d_in[0];   // [B,N,D] f32
    const float* adj = (const float*)d_in[1];   // [B,N,N] f32
    const float* U   = (const float*)d_in[2];   // [D,D]   f32
    float* out = (float*)d_out;                 // [B,N,D] f32

    _Float16* xT = (_Float16*)d_ws;                       // [B][D][N] f16, 4 MB
    _Float16* UT = xT + (size_t)Bb * Dd * Nn;             // [D][D]    f16, 32 KB

    dim3 tb(32, 8);
    transpose_f32_to_f16<<<dim3(Dd / 32, Nn / 32, Bb), tb, 0, stream>>>(x, xT, Nn, Dd);
    transpose_f32_to_f16<<<dim3(Dd / 32, Dd / 32, 1), tb, 0, stream>>>(U, UT, Dd, Dd);

    gcn_fused<<<dim3((Bb * Nn / 16) / 8), dim3(256), 0, stream>>>(adj, xT, UT, out);
}
